// GroupedQueryAttention_78065325572106
// MI455X (gfx1250) — matrix operand order repaired and checked by reading, not compiled, not run on hardware
//
#include <hip/hip_runtime.h>

// ---------------------------------------------------------------------------
// GQA forward for MI455X (gfx1250, wave32, WMMA 16x16x32 bf16)
// ---------------------------------------------------------------------------

#define S_LEN   2048
#define BATCH   2
#define ROWS    (BATCH * S_LEN)      // 4096
#define D_IN    2048
#define NHEADS  16
#define NKV     4
#define HDIM    128
#define D_OUT   2048
#define NQKV    3072                 // 2048 q + 512 k + 512 v

typedef __attribute__((ext_vector_type(16))) __bf16          v16bf_t;
typedef __attribute__((ext_vector_type(16))) unsigned short  v16u_t;
typedef __attribute__((ext_vector_type(8)))  unsigned short  v8u_t;
typedef __attribute__((ext_vector_type(8)))  float           v8f_t;

union BF16Tile { v16u_t u; v16bf_t b; };

__device__ __forceinline__ unsigned short f2bf(float f) {
  unsigned int u = __float_as_uint(f);
  u += 0x7FFFu + ((u >> 16) & 1u);   // round-to-nearest-even
  return (unsigned short)(u >> 16);
}

// A matrix (16x32 bf16, MxK), row-major source with leading dim lda.
// Lane L: row M = L&15 ; K = (i<8 ? i : i+8) + (L>=16 ? 8 : 0)
__device__ __forceinline__ v16bf_t load_A(const unsigned short* src, int lda, int lane) {
  int m    = lane & 15;
  int base = (lane & 16) >> 1;                 // 0 or 8
  const unsigned short* p = src + m * lda + base;
  v8u_t lo = *(const v8u_t*)(p);               // K = base+0..7
  v8u_t hi = *(const v8u_t*)(p + 16);          // K = base+16..23
  BF16Tile t;
#pragma unroll
  for (int i = 0; i < 8; ++i) { t.u[i] = lo[i]; t.u[i + 8] = hi[i]; }
  return t.b;
}

// B matrix (32x16 bf16, KxN), row-major [K][N] source: lane column N = L&15,
// K = (i<8 ? i : i+8) + (L>=16 ? 8 : 0).  Caller pre-offsets src by k0*ldb + n0.
__device__ __forceinline__ v16bf_t load_B(const unsigned short* src, int ldb, int lane) {
  BF16Tile t;
  int n    = lane & 15;
  int base = (lane & 16) >> 1;
#pragma unroll
  for (int i = 0; i < 16; ++i) {
    int k = (i < 8 ? i : i + 8) + base;
    t.u[i] = src[(size_t)k * ldb + n];
  }
  return t.b;
}

__device__ __forceinline__ v8f_t wmma_bf(v16bf_t a, v16bf_t b, v8f_t c) {
  return __builtin_amdgcn_wmma_f32_16x16x32_bf16(false, a, false, b, (short)0, c,
                                                 false, false);
}

// ---------------------------------------------------------------------------
// One wave computes a 32x64 f32 tile of C = A(bf16) * B(bf16).
// Two M-subtiles share every B fragment (8 WMMAs / 192B of loads).  Both
// operand streams are double-buffered one k-iteration ahead using moving
// pointers (add-only address update), with a peeled final compute block so no
// loads run past the buffers.  No in-loop prefetch: the buffered loads already
// give a full iteration of latency distance, and the extra address chains
// were spilling to scratch.
// ---------------------------------------------------------------------------
__device__ __forceinline__ void gemm_wave_32x64(const unsigned short* A, int lda,
                                                const unsigned short* B, int ldb,
                                                float* C, int ldc, int K, int lane) {
  v8f_t acc0[4] = {};
  v8f_t acc1[4] = {};

  const unsigned short* pA = A;
  const unsigned short* pB = B;
  const size_t astep16 = (size_t)16 * lda;
  const size_t bstep   = (size_t)32 * ldb;

  v16bf_t a0 = load_A(pA, lda, lane);
  v16bf_t a1 = load_A(pA + astep16, lda, lane);
  v16bf_t b[4];
#pragma unroll
  for (int j = 0; j < 4; ++j) b[j] = load_B(pB + j * 16, ldb, lane);

  int nk = K >> 5;
  for (int it = 0; it < nk - 1; ++it) {
    pA += 32;
    pB += bstep;
    v16bf_t a0n = load_A(pA, lda, lane);
    v16bf_t a1n = load_A(pA + astep16, lda, lane);
    v16bf_t bn[4];
#pragma unroll
    for (int j = 0; j < 4; ++j) bn[j] = load_B(pB + j * 16, ldb, lane);

#pragma unroll
    for (int j = 0; j < 4; ++j) {
      acc0[j] = wmma_bf(a0, b[j], acc0[j]);
      acc1[j] = wmma_bf(a1, b[j], acc1[j]);
    }
    a0 = a0n; a1 = a1n;
#pragma unroll
    for (int j = 0; j < 4; ++j) b[j] = bn[j];
  }
  // peeled last k-step: consume the final pre-loaded fragments
#pragma unroll
  for (int j = 0; j < 4; ++j) {
    acc0[j] = wmma_bf(a0, b[j], acc0[j]);
    acc1[j] = wmma_bf(a1, b[j], acc1[j]);
  }

  int half = lane >> 4, n = lane & 15;
#pragma unroll
  for (int j = 0; j < 4; ++j)
#pragma unroll
    for (int v = 0; v < 8; ++v) {
      C[(size_t)(v + 8 * half) * ldc + j * 16 + n]      = acc0[j][v];
      C[(size_t)(16 + v + 8 * half) * ldc + j * 16 + n] = acc1[j][v];
    }
}

// ---------------------------------------------------------------------------
// Kernel 1: fp32 -> bf16 convert (4 elements / thread)
// ---------------------------------------------------------------------------
__global__ void cvt_bf16_kernel(const float* __restrict__ src,
                                unsigned short* __restrict__ dst, int n4) {
  int i = blockIdx.x * blockDim.x + threadIdx.x;
  if (i < n4) {
    float4 f = ((const float4*)src)[i];
    ushort4 o;
    o.x = f2bf(f.x); o.y = f2bf(f.y); o.z = f2bf(f.z); o.w = f2bf(f.w);
    ((ushort4*)dst)[i] = o;
  }
}

// ---------------------------------------------------------------------------
// Kernel 2: fused QKV projection.  qkv[4096][3072] = xb[4096][2048] @ [Wq|Wk|Wv]
// ---------------------------------------------------------------------------
__global__ void qkv_gemm_kernel(const unsigned short* __restrict__ xb,
                                const unsigned short* __restrict__ wqb,
                                const unsigned short* __restrict__ wkb,
                                const unsigned short* __restrict__ wvb,
                                float* __restrict__ qkv) {
  int lane = threadIdx.x & 31;
  int wid  = (blockIdx.x * blockDim.x + threadIdx.x) >> 5;
  const int NG = NQKV / 64;                    // 48
  int mi = wid / NG, ng = wid % NG;            // mi in [0,128): 32-row tiles
  int n0 = ng * 64;
  const unsigned short* B; int ldb, bn;
  if (n0 < 2048)      { B = wqb; ldb = 2048; bn = n0; }
  else if (n0 < 2560) { B = wkb; ldb = 512;  bn = n0 - 2048; }
  else                { B = wvb; ldb = 512;  bn = n0 - 2560; }
  const unsigned short* Abase = xb + (size_t)mi * 32 * D_IN;
  __builtin_prefetch(Abase, 0, 1);             // one-shot global_prefetch_b8
  __builtin_prefetch(B + bn, 0, 1);
  gemm_wave_32x64(Abase, D_IN,
                  B + bn, ldb,
                  qkv + (size_t)mi * 32 * NQKV + n0, NQKV, D_IN, lane);
}

// ---------------------------------------------------------------------------
// Kernel 3: per-head RMSNorm + RoPE; also v fp32->bf16 relayout.
//   q  -> qb  [b][h][s][128]   bf16
//   k  -> ktb [b][g][128][s]   bf16 (transposed for QK^T B-operand)
//   v  -> vb  [b][g][s][128]   bf16
// One wave per row; lane owns elements {lane, lane+32, lane+64, lane+96}.
// ---------------------------------------------------------------------------
__global__ void norm_rope_kernel(const float* __restrict__ qkv,
                                 const float* __restrict__ cosb,
                                 const float* __restrict__ sinb,
                                 const float* __restrict__ qn,
                                 const float* __restrict__ kn,
                                 unsigned short* __restrict__ qb,
                                 unsigned short* __restrict__ ktb,
                                 unsigned short* __restrict__ vb) {
  int lane = threadIdx.x & 31;
  int wid  = (blockIdx.x * blockDim.x + threadIdx.x) >> 5;

  if (wid < BATCH * NHEADS * S_LEN) {                       // ---- Q ----
    int b = wid >> 15, rem = wid & 32767, h = rem >> 11, pos = rem & 2047;
    const float* src = qkv + (size_t)(b * S_LEN + pos) * NQKV + h * HDIM;
    float y0 = src[lane], y1 = src[lane + 32], y2 = src[lane + 64], y3 = src[lane + 96];
    float ss = y0 * y0 + y1 * y1 + y2 * y2 + y3 * y3;
#pragma unroll
    for (int off = 16; off; off >>= 1) ss += __shfl_xor(ss, off, 32);
    float inv = rsqrtf(ss * (1.0f / 128.0f) + 1e-6f);
    y0 *= inv * qn[lane];      y1 *= inv * qn[lane + 32];
    y2 *= inv * qn[lane + 64]; y3 *= inv * qn[lane + 96];
    const float* cr = cosb + (size_t)pos * HDIM;
    const float* sr = sinb + (size_t)pos * HDIM;
    float o0 = y0 * cr[lane]      - y2 * sr[lane];
    float o1 = y1 * cr[lane + 32] - y3 * sr[lane + 32];
    float o2 = y2 * cr[lane + 64] + y0 * sr[lane + 64];
    float o3 = y3 * cr[lane + 96] + y1 * sr[lane + 96];
    unsigned short* dst = qb + ((size_t)((b * NHEADS + h) * S_LEN + pos)) * HDIM;
    dst[lane] = f2bf(o0); dst[lane + 32] = f2bf(o1);
    dst[lane + 64] = f2bf(o2); dst[lane + 96] = f2bf(o3);
  } else if (wid < BATCH * NHEADS * S_LEN + BATCH * NKV * S_LEN) {   // ---- K ----
    int w = wid - BATCH * NHEADS * S_LEN;
    int b = w >> 13, rem = w & 8191, g = rem >> 11, pos = rem & 2047;
    const float* src = qkv + (size_t)(b * S_LEN + pos) * NQKV + 2048 + g * HDIM;
    float y0 = src[lane], y1 = src[lane + 32], y2 = src[lane + 64], y3 = src[lane + 96];
    float ss = y0 * y0 + y1 * y1 + y2 * y2 + y3 * y3;
#pragma unroll
    for (int off = 16; off; off >>= 1) ss += __shfl_xor(ss, off, 32);
    float inv = rsqrtf(ss * (1.0f / 128.0f) + 1e-6f);
    y0 *= inv * kn[lane];      y1 *= inv * kn[lane + 32];
    y2 *= inv * kn[lane + 64]; y3 *= inv * kn[lane + 96];
    const float* cr = cosb + (size_t)pos * HDIM;
    const float* sr = sinb + (size_t)pos * HDIM;
    float o0 = y0 * cr[lane]      - y2 * sr[lane];
    float o1 = y1 * cr[lane + 32] - y3 * sr[lane + 32];
    float o2 = y2 * cr[lane + 64] + y0 * sr[lane + 64];
    float o3 = y3 * cr[lane + 96] + y1 * sr[lane + 96];
    unsigned short* dst = ktb + (size_t)(b * NKV + g) * HDIM * S_LEN;
    dst[(size_t)(lane)      * S_LEN + pos] = f2bf(o0);
    dst[(size_t)(lane + 32) * S_LEN + pos] = f2bf(o1);
    dst[(size_t)(lane + 64) * S_LEN + pos] = f2bf(o2);
    dst[(size_t)(lane + 96) * S_LEN + pos] = f2bf(o3);
  } else {                                                   // ---- V ----
    int w = wid - (BATCH * NHEADS * S_LEN + BATCH * NKV * S_LEN);
    int b = w >> 13, rem = w & 8191, g = rem >> 11, pos = rem & 2047;
    const float* src = qkv + (size_t)(b * S_LEN + pos) * NQKV + 2560 + g * HDIM;
    unsigned short* dst = vb + ((size_t)((b * NKV + g) * S_LEN + pos)) * HDIM;
    dst[lane]      = f2bf(src[lane]);
    dst[lane + 32] = f2bf(src[lane + 32]);
    dst[lane + 64] = f2bf(src[lane + 64]);
    dst[lane + 96] = f2bf(src[lane + 96]);
  }
}

// ---------------------------------------------------------------------------
// Kernel 4: causal flash attention.  One wave per (b, h, 16-row q tile).
// ctxb[b][s][h*128+d] bf16 output, ready for the Wo GEMM.
// ---------------------------------------------------------------------------
__global__ void attention_kernel(const unsigned short* __restrict__ qb,
                                 const unsigned short* __restrict__ ktb,
                                 const unsigned short* __restrict__ vb,
                                 unsigned short* __restrict__ ctxb) {
  __shared__ float Pst[4][16 * 33];            // padded 16x32 P tile per wave
  int lane  = threadIdx.x & 31;
  int wslot = threadIdx.x >> 5;
  int wid   = (blockIdx.x * blockDim.x + threadIdx.x) >> 5;
  int b = wid >> 11, rem = wid & 2047, h = rem >> 7, qt = rem & 127;
  int g = h >> 2;                              // rep = NHEADS/NKV = 4
  int q0 = qt * 16;

  const unsigned short* qhead = qb  + ((size_t)(b * NHEADS + h) * S_LEN) * HDIM;
  const unsigned short* kt    = ktb + (size_t)(b * NKV + g) * HDIM * S_LEN;
  const unsigned short* vg    = vb  + (size_t)(b * NKV + g) * S_LEN * HDIM;

  v16bf_t qa[4];
#pragma unroll
  for (int c = 0; c < 4; ++c)
    qa[c] = load_A(qhead + (size_t)q0 * HDIM + c * 32, HDIM, lane);

  v8f_t ctx[8] = {};
  float mrow[8], lrow[8];
#pragma unroll
  for (int v = 0; v < 8; ++v) { mrow[v] = -1e30f; lrow[v] = 0.0f; }

  int half = lane >> 4, nn = lane & 15;
  float* P = Pst[wslot];
  const float scale = 0.08838834764831845f;    // 128^-0.5
  int kbmax = (q0 + 15) >> 5;                  // inclusive causal block limit

  for (int kb = 0; kb <= kbmax; ++kb) {
    int kpos0 = kb * 32;
    v8f_t s0 = {}, s1 = {};
#pragma unroll
    for (int c = 0; c < 4; ++c) {              // reduce over d = 128
      const unsigned short* kbase = kt + (size_t)(c * 32) * S_LEN + kpos0;
      v16bf_t b0 = load_B(kbase,      S_LEN, lane);
      v16bf_t b1 = load_B(kbase + 16, S_LEN, lane);
      s0 = wmma_bf(qa[c], b0, s0);
      s1 = wmma_bf(qa[c], b1, s1);
    }
    // scale + causal mask + online softmax (row stats live per (v, lane-half))
#pragma unroll
    for (int v = 0; v < 8; ++v) {
      int qpos = q0 + v + 8 * half;
      float a  = s0[v] * scale;
      float bb = s1[v] * scale;
      if (kpos0 + nn      > qpos) a  = -1e30f;
      if (kpos0 + 16 + nn > qpos) bb = -1e30f;
      float t = fmaxf(a, bb);
#pragma unroll
      for (int off = 1; off < 16; off <<= 1) t = fmaxf(t, __shfl_xor(t, off, 32));
      float mn  = fmaxf(mrow[v], t);
      float cor = __expf(mrow[v] - mn);
      mrow[v] = mn;
      float p0 = __expf(a - mn), p1 = __expf(bb - mn);
      float rs = p0 + p1;
#pragma unroll
      for (int off = 1; off < 16; off <<= 1) rs += __shfl_xor(rs, off, 32);
      lrow[v] = lrow[v] * cor + rs;
#pragma unroll
      for (int d8 = 0; d8 < 8; ++d8) ctx[d8][v] *= cor;
      int m = v + 8 * half;
      P[m * 33 + nn]      = p0;                // stage P (C layout) in LDS
      P[m * 33 + 16 + nn] = p1;
    }
    __asm__ volatile("s_wait_dscnt 0" ::: "memory");
    // re-read P from LDS in WMMA A layout, converting to bf16
    BF16Tile pa;
    {
      int m    = lane & 15;
      int base = (lane & 16) >> 1;
#pragma unroll
      for (int i = 0; i < 16; ++i) {
        int K = (i < 8 ? i : i + 8) + base;
        pa.u[i] = f2bf(P[m * 33 + K]);
      }
    }
    __asm__ volatile("s_wait_dscnt 0" ::: "memory");
#pragma unroll
    for (int d8 = 0; d8 < 8; ++d8) {           // ctx += P @ V
      v16bf_t bv = load_B(vg + (size_t)kpos0 * HDIM + d8 * 16, HDIM, lane);
      ctx[d8] = wmma_bf(pa.b, bv, ctx[d8]);
    }
  }

  unsigned short* dst = ctxb + ((size_t)(b * S_LEN + q0)) * D_OUT + h * HDIM;
#pragma unroll
  for (int v = 0; v < 8; ++v) {
    float invl = 1.0f / lrow[v];
    int m = v + 8 * half;
#pragma unroll
    for (int d8 = 0; d8 < 8; ++d8)
      dst[(size_t)m * D_OUT + d8 * 16 + nn] = f2bf(ctx[d8][v] * invl);
  }
}

// ---------------------------------------------------------------------------
// Kernel 5: output projection  out[4096][2048] = ctxb @ Wo
// ---------------------------------------------------------------------------
__global__ void out_gemm_kernel(const unsigned short* __restrict__ ctxb,
                                const unsigned short* __restrict__ wob,
                                float* __restrict__ out) {
  int lane = threadIdx.x & 31;
  int wid  = (blockIdx.x * blockDim.x + threadIdx.x) >> 5;
  const int NG = D_IN / 64;                    // 32
  int mi = wid / NG, ng = wid % NG;            // mi in [0,128): 32-row tiles
  const unsigned short* Abase = ctxb + (size_t)mi * 32 * D_OUT;
  __builtin_prefetch(Abase, 0, 1);             // one-shot global_prefetch_b8
  __builtin_prefetch(wob + ng * 64, 0, 1);
  gemm_wave_32x64(Abase, D_OUT,
                  wob + ng * 64, D_IN,
                  out + (size_t)mi * 32 * D_IN + ng * 64, D_IN, D_OUT, lane);
}

// ---------------------------------------------------------------------------
// Host side
// ---------------------------------------------------------------------------
extern "C" void kernel_launch(void* const* d_in, const int* in_sizes, int n_in,
                              void* d_out, int out_size, void* d_ws, size_t ws_size,
                              hipStream_t stream) {
  (void)in_sizes; (void)n_in; (void)out_size; (void)ws_size;
  const float* x   = (const float*)d_in[0];
  // d_in[1] = mask (unused; causal mask computed analytically)
  const float* cosb = (const float*)d_in[2];
  const float* sinb = (const float*)d_in[3];
  const float* Wq  = (const float*)d_in[4];
  const float* Wk  = (const float*)d_in[5];
  const float* Wv  = (const float*)d_in[6];
  const float* Wo  = (const float*)d_in[7];
  const float* qn  = (const float*)d_in[8];
  const float* kn  = (const float*)d_in[9];
  float* out = (float*)d_out;

  char* ws = (char*)d_ws;
  const size_t MB = 1024ull * 1024ull;
  unsigned short* xb   = (unsigned short*)(ws + 0);          // 16 MB
  unsigned short* wqb  = (unsigned short*)(ws + 16 * MB);    //  8 MB
  unsigned short* wkb  = (unsigned short*)(ws + 24 * MB);    //  2 MB
  unsigned short* wvb  = (unsigned short*)(ws + 26 * MB);    //  2 MB
  unsigned short* wob  = (unsigned short*)(ws + 28 * MB);    //  8 MB
  float*          qkv  = (float*)        (ws + 36 * MB);     // 48 MB
  unsigned short* qb   = (unsigned short*)(ws + 84 * MB);    // 16 MB
  unsigned short* ktb  = (unsigned short*)(ws + 100 * MB);   //  4 MB
  unsigned short* vb   = (unsigned short*)(ws + 104 * MB);   //  4 MB
  unsigned short* ctxb = (unsigned short*)(ws + 108 * MB);   // 16 MB

  // 1) converts
  auto cvt = [&](const float* s, unsigned short* d, int n) {
    int n4 = n / 4;
    cvt_bf16_kernel<<<(n4 + 255) / 256, 256, 0, stream>>>(s, d, n4);
  };
  cvt(x,  xb,  ROWS * D_IN);
  cvt(Wq, wqb, D_IN * D_OUT);
  cvt(Wk, wkb, D_IN * NKV * HDIM);
  cvt(Wv, wvb, D_IN * NKV * HDIM);
  cvt(Wo, wob, D_OUT * D_IN);

  // 2) QKV projection: 128 M-tiles (32 rows) * 48 N-groups = 6144 waves
  qkv_gemm_kernel<<<1536, 128, 0, stream>>>(xb, wqb, wkb, wvb, qkv);

  // 3) RMSNorm + RoPE + relayout: 98304 waves
  norm_rope_kernel<<<24576, 128, 0, stream>>>(qkv, cosb, sinb, qn, kn, qb, ktb, vb);

  // 4) attention: 2*16*128 = 4096 waves
  attention_kernel<<<1024, 128, 0, stream>>>(qb, ktb, vb, ctxb);

  // 5) output projection: 128 * 32 = 4096 waves
  out_gemm_kernel<<<1024, 128, 0, stream>>>(ctxb, wob, out);
}
